// SpoTMamba_25555055411709
// MI455X (gfx1250) — matrix operand order repaired and checked by reading, not compiled, not run on hardware
//
#include <hip/hip_runtime.h>
#include <hip/hip_bf16.h>

// ---------------- model constants ----------------
#define kE     32
#define kTOTW  9
#define kLW    8
#define kDW    864      // DWALK = 9*3*32
#define kDC    128      // DSCAN
#define kFF    256
#define kNL    2
#define kN     207
#define kB     8
#define kT     12
#define kIN    3
#define kDS    8        // mamba state dim
#define kK     4        // conv kernel
#define kNH    4

typedef __attribute__((ext_vector_type(16))) _Float16 v16h;
typedef __attribute__((ext_vector_type(8)))  _Float16 v8h;
typedef __attribute__((ext_vector_type(2)))  _Float16 h2;
typedef __attribute__((ext_vector_type(8)))  float    v8f;

static inline int cdiv(int a, int b) { return (a + b - 1) / b; }

// ---------------- device helpers ----------------
__device__ __forceinline__ float siluf(float x) { return x / (1.f + __expf(-x)); }
__device__ __forceinline__ float geluf(float x) {
  const float c = 0.7978845608028654f;
  return 0.5f * x * (1.f + tanhf(c * (x + 0.044715f * x * x * x)));
}

enum { ACT_NONE = 0, ACT_SILU = 1, ACT_GELU = 2, ACT_RELU = 3, ACT_SOFTPLUS = 4 };

template <int ACT> __device__ __forceinline__ float act_apply(float x) {
  if constexpr (ACT == ACT_SILU)      return siluf(x);
  else if constexpr (ACT == ACT_GELU) return geluf(x);
  else if constexpr (ACT == ACT_RELU) return fmaxf(x, 0.f);
  else if constexpr (ACT == ACT_SOFTPLUS) return (x > 20.f) ? x : log1pf(__expf(x));
  else return x;
}

// ---------------- WMMA GEMM:  C = act(A[MxK] @ B[KxN] + bias) ----------------
// 256 threads = 8 waves; workgroup tile 64(M) x 64(N); each wave a 16x32 strip
// = two v_wmma_f32_16x16x32_f16 sharing one A fragment. Double-buffered LDS,
// B stored transposed so fragments are contiguous (ds_load_b128). Staging is
// split into a batched-load phase and a convert/store phase so global loads
// stay in flight together. Fragments use contiguous k order (k = 16*(lane/16) + s) for A and B.
#define TM 64
#define TN 64
#define TK 32
#define TKP (TK + 8)   // padded row stride in halves: 80B, 16B-aligned

template <int ACT>
__global__ void __launch_bounds__(256)
gemm_wmma(const float* __restrict__ A, int lda,
          const float* __restrict__ B, int ldb,
          const float* __restrict__ bias,
          float* __restrict__ C, int ldc,
          int M, int N, int K) {
  __shared__ __attribute__((aligned(16))) _Float16 As [2][TM][TKP];
  __shared__ __attribute__((aligned(16))) _Float16 BsT[2][TN][TKP];

  const int tid  = threadIdx.x;
  const int lane = tid & 31;
  const int wave = tid >> 5;   // 0..7
  const int wm   = wave & 3;   // 4 M-subtiles (16 rows each)
  const int wn   = wave >> 2;  // 2 N-strips (32 cols each)
  const int m0   = blockIdx.y * TM;
  const int n0   = blockIdx.x * TN;

  const bool fastA = (m0 + TM <= M) && ((lda & 1) == 0);
  const bool fastB = (n0 + TN <= N) && ((ldb & 1) == 0);
  const int  nk    = (K + TK - 1) / TK;

  auto stage = [&](int kt, int buf) {
    const int  k0    = kt * TK;
    const bool kfull = (k0 + TK <= K);
    // ---- A tile 64x32 ----
    if (fastA && kfull) {
      float2 fa[4];
#pragma unroll
      for (int i = 0; i < 4; ++i) {
        int p = tid + i * 256, r = p >> 4, cp = (p & 15) << 1;
        fa[i] = *(const float2*)(A + (size_t)(m0 + r) * lda + k0 + cp);
      }
#pragma unroll
      for (int i = 0; i < 4; ++i) {
        int p = tid + i * 256, r = p >> 4, cp = (p & 15) << 1;
        h2 hv = { (_Float16)fa[i].x, (_Float16)fa[i].y };
        *(h2*)&As[buf][r][cp] = hv;
      }
    } else {
#pragma unroll
      for (int i = 0; i < 8; ++i) {
        int idx = tid + i * 256;
        int r = idx >> 5, c = idx & 31;
        int gm = m0 + r, gk = k0 + c;
        float v = (gm < M && gk < K) ? A[(size_t)gm * lda + gk] : 0.f;
        As[buf][r][c] = (_Float16)v;
      }
    }
    // ---- B tile 32x64 (stored transposed) ----
    if (fastB && kfull) {
      float2 fb[4];
#pragma unroll
      for (int i = 0; i < 4; ++i) {
        int p = tid + i * 256, r = p >> 5, cp = (p & 31) << 1;
        fb[i] = *(const float2*)(B + (size_t)(k0 + r) * ldb + n0 + cp);
      }
#pragma unroll
      for (int i = 0; i < 4; ++i) {
        int p = tid + i * 256, r = p >> 5, cp = (p & 31) << 1;
        BsT[buf][cp][r]     = (_Float16)fb[i].x;
        BsT[buf][cp + 1][r] = (_Float16)fb[i].y;
      }
    } else {
#pragma unroll
      for (int i = 0; i < 8; ++i) {
        int idx = tid + i * 256;
        int r = idx >> 6, c = idx & 63;
        int gk = k0 + r, gn = n0 + c;
        float v = (gk < K && gn < N) ? B[(size_t)gk * ldb + gn] : 0.f;
        BsT[buf][c][r] = (_Float16)v;
      }
    }
    // prefetch the K+2 A slab into GL2 (global_prefetch_b8)
    {
      int pk = k0 + 2 * TK;
      int pm = m0 + (tid & 63);
      if (pk < K && pm < M) __builtin_prefetch(&A[(size_t)pm * lda + pk], 0, 1);
    }
  };

  v8f acc0 = {}, acc1 = {};
  stage(0, 0);

  // per-lane fragment addresses (contiguous k order, same for A and B)
  const int am  = wm * 16 + (lane & 15);
  const int akg = (lane >> 4) << 4;   // 0/16: lanes 0-15 hold K 0-15; lanes 16-31 hold 16-31
  const int bn0 = wn * 32 + (lane & 15);
  const int bn1 = bn0 + 16;
  const int bkh = (lane >> 4) << 4;   // 0/16: lanes 0-15 hold K 0-15; lanes 16-31 hold 16-31

  for (int kt = 0; kt < nk; ++kt) {
    __syncthreads();
    if (kt + 1 < nk) stage(kt + 1, (kt + 1) & 1);
    const int buf = kt & 1;
    v8h alo  = *(const v8h*)&As [buf][am][akg];        // ds_load_b128
    v8h ahi  = *(const v8h*)&As [buf][am][akg + 8];
    v8h blo0 = *(const v8h*)&BsT[buf][bn0][bkh];
    v8h bhi0 = *(const v8h*)&BsT[buf][bn0][bkh + 8];
    v8h blo1 = *(const v8h*)&BsT[buf][bn1][bkh];
    v8h bhi1 = *(const v8h*)&BsT[buf][bn1][bkh + 8];
    v16h af  = __builtin_shufflevector(alo,  ahi,  0,1,2,3,4,5,6,7,8,9,10,11,12,13,14,15);
    v16h bf0 = __builtin_shufflevector(blo0, bhi0, 0,1,2,3,4,5,6,7,8,9,10,11,12,13,14,15);
    v16h bf1 = __builtin_shufflevector(blo1, bhi1, 0,1,2,3,4,5,6,7,8,9,10,11,12,13,14,15);
    acc0 = __builtin_amdgcn_wmma_f32_16x16x32_f16(false, af, false, bf0, (short)0, acc0, false, false);
    acc1 = __builtin_amdgcn_wmma_f32_16x16x32_f16(false, af, false, bf1, (short)0, acc1, false, false);
  }

  // store: C/D layout — VGPR r holds row (lane<16 ? r : 8+r), col = lane%16
  const int col  = lane & 15;
  const int half = lane >> 4;
  const int gn0  = n0 + wn * 32 + col;
  const int gn1  = gn0 + 16;
  if (gn0 < N) {
    const float bv = bias ? bias[gn0] : 0.f;
#pragma unroll
    for (int r = 0; r < 8; ++r) {
      int gm = m0 + wm * 16 + half * 8 + r;
      if (gm < M) C[(size_t)gm * ldc + gn0] = act_apply<ACT>(acc0[r] + bv);
    }
  }
  if (gn1 < N) {
    const float bv = bias ? bias[gn1] : 0.f;
#pragma unroll
    for (int r = 0; r < 8; ++r) {
      int gm = m0 + wm * 16 + half * 8 + r;
      if (gm < M) C[(size_t)gm * ldc + gn1] = act_apply<ACT>(acc1[r] + bv);
    }
  }
}

// ---------------- elementwise / special kernels ----------------

__global__ void embed_walk_kernel(const int* __restrict__ seq, const float* __restrict__ cost,
                                  const float* __restrict__ node_table,
                                  const float* __restrict__ cost_w, const float* __restrict__ cost_b,
                                  const float* __restrict__ pos_table, float* __restrict__ out) {
  int tid = blockIdx.x * blockDim.x + threadIdx.x;
  const int total = kN * kLW * kTOTW * kE;
  if (tid >= total) return;
  int e = tid & 31;
  int r = tid >> 5;
  int w = r % kTOTW; r /= kTOTW;
  int l = r % kLW;
  int n = r / kLW;
  int flat = (n * kLW + l) * kTOTW + w;
  float* o = out + (size_t)(n * kLW + l) * kDW + w * 3 * kE;
  o[e]          = node_table[seq[flat] * kE + e];
  o[kE + e]     = cost[flat] * cost_w[e] + cost_b[e];
  o[2 * kE + e] = pos_table[l * kE + e];
}

// per-row layernorm, optional fused residual add (v = x + res)
__global__ void __launch_bounds__(256)
ln_kernel(const float* __restrict__ x, const float* __restrict__ res,
          const float* __restrict__ g, const float* __restrict__ b,
          float* __restrict__ out, int D) {
  __shared__ float sred[256];
  const int row = blockIdx.x;
  const float* xr = x + (size_t)row * D;
  const float* rr = res ? res + (size_t)row * D : nullptr;
  float vals[4];
  int cnt = 0;
  float s = 0.f;
  for (int i = threadIdx.x; i < D; i += 256) {
    float v = xr[i] + (rr ? rr[i] : 0.f);
    vals[cnt++] = v;
    s += v;
  }
  sred[threadIdx.x] = s; __syncthreads();
  for (int o = 128; o > 0; o >>= 1) { if (threadIdx.x < o) sred[threadIdx.x] += sred[threadIdx.x + o]; __syncthreads(); }
  const float mean = sred[0] / (float)D; __syncthreads();
  float s2 = 0.f; cnt = 0;
  for (int i = threadIdx.x; i < D; i += 256) { float d = vals[cnt++] - mean; s2 += d * d; }
  sred[threadIdx.x] = s2; __syncthreads();
  for (int o = 128; o > 0; o >>= 1) { if (threadIdx.x < o) sred[threadIdx.x] += sred[threadIdx.x + o]; __syncthreads(); }
  const float inv = rsqrtf(sred[0] / (float)D + 1e-5f);
  cnt = 0;
  for (int i = threadIdx.x; i < D; i += 256)
    out[(size_t)row * D + i] = (vals[cnt++] - mean) * inv * g[i] + b[i];
}

// depthwise causal conv (K=4) over L + silu; xd = first Di cols of xz (row stride ldx)
__global__ void mamba_conv_kernel(const float* __restrict__ xz, int ldx,
                                  const float* __restrict__ conv_w, const float* __restrict__ conv_b,
                                  float* __restrict__ xc, int Bt, int L, int Di) {
  int tid = blockIdx.x * blockDim.x + threadIdx.x;
  if (tid >= Bt * Di) return;
  int di = tid % Di, b = tid / Di;
  float w0 = conv_w[0 * Di + di], w1 = conv_w[1 * Di + di];
  float w2 = conv_w[2 * Di + di], w3 = conv_w[3 * Di + di];
  float cb = conv_b[di];
  const float* xd = xz + (size_t)b * L * ldx + di;
  float x0 = 0.f, x1 = 0.f, x2 = 0.f;
  for (int t = 0; t < L; ++t) {
    float x3 = xd[(size_t)t * ldx];
    float v = x0 * w0 + x1 * w1 + x2 * w2 + x3 * w3 + cb;
    xc[(size_t)(b * L + t) * Di + di] = siluf(v);
    x0 = x1; x1 = x2; x2 = x3;
  }
}

// selective scan: h = exp(dt*A)*h + dt*Bm*xc ; y = (sum_s h_s*Cm_s + xc*D) * silu(z)
__global__ void mamba_scan_kernel(const float* __restrict__ dt,        // already softplus(+dt_b)
                                  const float* __restrict__ A_log, const float* __restrict__ Dv,
                                  const float* __restrict__ dbc, int lddbc, int boff, int coff,
                                  const float* __restrict__ xc,
                                  const float* __restrict__ xz, int ldz, int zoff,
                                  float* __restrict__ y, int Bt, int L, int Di) {
  int tid = blockIdx.x * blockDim.x + threadIdx.x;
  if (tid >= Bt * Di) return;
  int di = tid % Di, b = tid / Di;
  float Ac[kDS];
#pragma unroll
  for (int s = 0; s < kDS; ++s) Ac[s] = -__expf(A_log[(size_t)di * kDS + s]);
  float h[kDS];
#pragma unroll
  for (int s = 0; s < kDS; ++s) h[s] = 0.f;
  const float Dp = Dv[di];
  for (int t = 0; t < L; ++t) {
    size_t tok = (size_t)(b * L + t);
    float dtv = dt[tok * Di + di];
    float xcv = xc[tok * Di + di];
    const float* row = dbc + tok * lddbc;
    float accv = 0.f;
#pragma unroll
    for (int s = 0; s < kDS; ++s) {
      float a = __expf(dtv * Ac[s]);
      h[s] = a * h[s] + dtv * row[boff + s] * xcv;
      accv += h[s] * row[coff + s];
    }
    float zv = xz[tok * ldz + zoff + di];
    y[tok * Di + di] = (accv + xcv * Dp) * siluf(zv);
  }
}

__global__ void flip_time_kernel(const float* __restrict__ in, float* __restrict__ out,
                                 int Bt, int L, int D) {
  int tid = blockIdx.x * blockDim.x + threadIdx.x;
  if (tid >= Bt * L * D) return;
  int d = tid % D;
  int t = (tid / D) % L;
  int b = tid / (D * L);
  out[((size_t)(b * L + (L - 1 - t))) * D + d] = in[tid];
}

__global__ void copy_kernel(float* __restrict__ a, const float* __restrict__ b, int n) {
  int i = blockIdx.x * blockDim.x + threadIdx.x;
  if (i < n) a[i] = b[i];
}
__global__ void add_kernel(float* __restrict__ a, const float* __restrict__ b, int n) {
  int i = blockIdx.x * blockDim.x + threadIdx.x;
  if (i < n) a[i] += b[i];
}
__global__ void gelu_kernel(const float* __restrict__ in, float* __restrict__ out, int n) {
  int i = blockIdx.x * blockDim.x + threadIdx.x;
  if (i < n) out[i] = geluf(in[i]);
}

// h_walk_pre[n,d] = sum_l gelu(h[n,l,d]) * conv1_w[l] + conv1_b
__global__ void walk_pool_kernel(const float* __restrict__ h, const float* __restrict__ w8,
                                 const float* __restrict__ b0, float* __restrict__ out) {
  int tid = blockIdx.x * blockDim.x + threadIdx.x;
  if (tid >= kN * kDW) return;
  int d = tid % kDW, n = tid / kDW;
  float s = 0.f;
#pragma unroll
  for (int l = 0; l < kLW; ++l) s += geluf(h[((size_t)(n * kLW + l)) * kDW + d]) * w8[l];
  out[(size_t)n * kDW + d] = s + b0[0];
}

// build temporal tokens directly in (b*N+n, T, 128) layout
__global__ void temporal_embed_kernel(const float* __restrict__ x, const int* __restrict__ tod,
                                      const int* __restrict__ dow,
                                      const float* __restrict__ te_w, const float* __restrict__ te_b,
                                      const float* __restrict__ tod_table, const float* __restrict__ dow_table,
                                      const float* __restrict__ h_walk, float* __restrict__ ht) {
  int tid = blockIdx.x * blockDim.x + threadIdx.x;
  const int total = kB * kT * kN * kDC;
  if (tid >= total) return;
  int e = tid & 127;
  int n = (tid >> 7) % kN;
  int t = (tid / (kDC * kN)) % kT;
  int b = tid / (kDC * kN * kT);
  int btn = (b * kT + t) * kN + n;
  int j = e & 31, c = e >> 5;
  float v;
  if (c == 0) {
    const float* xr = x + (size_t)btn * kIN;
    v = xr[0] * te_w[j] + xr[1] * te_w[32 + j] + xr[2] * te_w[64 + j] + te_b[j];
  } else if (c == 1) v = tod_table[tod[btn] * kE + j];
  else if (c == 2)   v = dow_table[dow[btn] * kE + j];
  else               v = h_walk[n * kE + j];
  ht[((size_t)(b * kN + n) * kT + t) * kDC + e] = v;
}

__global__ void ht_to_hs_kernel(const float* __restrict__ ht, float* __restrict__ hs) {
  int tid = blockIdx.x * blockDim.x + threadIdx.x;
  const int total = kB * kT * kN * kDC;
  if (tid >= total) return;
  int e = tid & 127;
  int t = (tid >> 7) % kT;
  int n = (tid / (kDC * kT)) % kN;
  int b = tid / (kDC * kT * kN);
  hs[((size_t)(b * kT + t) * kN + n) * kDC + e] = ht[((size_t)(b * kN + n) * kT + t) * kDC + e];
}

// one block per (bt, head, query row): scores in LDS, softmax, weighted V
__global__ void __launch_bounds__(256)
attn_row_kernel(const float* __restrict__ qkv, float* __restrict__ out) {
  __shared__ float q[32];
  __shared__ float sc[kN];
  __shared__ float red[256];
  int bid = blockIdx.x;
  int l  = bid % kN;
  int hh = (bid / kN) % kNH;
  int bt = bid / (kN * kNH);
  const float* base = qkv + (size_t)bt * kN * 3 * kDC;
  if (threadIdx.x < 32) q[threadIdx.x] = base[(size_t)l * 384 + hh * 32 + threadIdx.x];
  __syncthreads();
  float lmax = -1e30f;
  for (int m = threadIdx.x; m < kN; m += 256) {
    const float* kr = base + (size_t)m * 384 + kDC + hh * 32;
    float s = 0.f;
#pragma unroll
    for (int d = 0; d < 32; ++d) s += q[d] * kr[d];
    s *= 0.17677669529663687f;  // 1/sqrt(32)
    sc[m] = s;
    lmax = fmaxf(lmax, s);
  }
  red[threadIdx.x] = lmax; __syncthreads();
  for (int o = 128; o > 0; o >>= 1) { if (threadIdx.x < o) red[threadIdx.x] = fmaxf(red[threadIdx.x], red[threadIdx.x + o]); __syncthreads(); }
  float mx = red[0]; __syncthreads();
  float lsum = 0.f;
  for (int m = threadIdx.x; m < kN; m += 256) { float e = __expf(sc[m] - mx); sc[m] = e; lsum += e; }
  red[threadIdx.x] = lsum; __syncthreads();
  for (int o = 128; o > 0; o >>= 1) { if (threadIdx.x < o) red[threadIdx.x] += red[threadIdx.x + o]; __syncthreads(); }
  float inv = 1.f / red[0];
  __syncthreads();
  if (threadIdx.x < 32) {
    int d = threadIdx.x;
    float o = 0.f;
    for (int m = 0; m < kN; ++m) o += sc[m] * base[(size_t)m * 384 + 2 * kDC + hh * 32 + d];
    out[((size_t)(bt * kN + l)) * kDC + hh * 32 + d] = o * inv;
  }
}

// ---------------- host side ----------------

static void gemm(hipStream_t s, const float* A, int lda, const float* B, int ldb,
                 const float* bias, float* C, int ldc, int M, int N, int K, int act) {
  dim3 grid(cdiv(N, TN), cdiv(M, TM)), blk(256);
  switch (act) {
    case ACT_SILU:     gemm_wmma<ACT_SILU><<<grid, blk, 0, s>>>(A, lda, B, ldb, bias, C, ldc, M, N, K); break;
    case ACT_GELU:     gemm_wmma<ACT_GELU><<<grid, blk, 0, s>>>(A, lda, B, ldb, bias, C, ldc, M, N, K); break;
    case ACT_RELU:     gemm_wmma<ACT_RELU><<<grid, blk, 0, s>>>(A, lda, B, ldb, bias, C, ldc, M, N, K); break;
    case ACT_SOFTPLUS: gemm_wmma<ACT_SOFTPLUS><<<grid, blk, 0, s>>>(A, lda, B, ldb, bias, C, ldc, M, N, K); break;
    default:           gemm_wmma<ACT_NONE><<<grid, blk, 0, s>>>(A, lda, B, ldb, bias, C, ldc, M, N, K); break;
  }
}

struct MambaP {
  const float *A_log, *D, *conv_b, *conv_w, *dt_b, *dt_w, *in_proj, *out_proj, *x_proj;
};

// out = mamba(hn); hn is (Bt*L, d)
static void run_mamba(hipStream_t s, const float* hn, float* out,
                      float* xz, float* xc, float* dbc, float* dtb, float* yb,
                      const MambaP& p, int Bt, int L, int d) {
  const int Di   = 2 * d;
  const int dtr  = (d + 15) / 16;
  const int Mtok = Bt * L;
  const int wdbc = dtr + 2 * kDS;
  gemm(s, hn, d, p.in_proj, 2 * Di, nullptr, xz, 2 * Di, Mtok, 2 * Di, d, ACT_NONE);
  int th = Bt * Di;
  mamba_conv_kernel<<<cdiv(th, 256), 256, 0, s>>>(xz, 2 * Di, p.conv_w, p.conv_b, xc, Bt, L, Di);
  gemm(s, xc, Di, p.x_proj, wdbc, nullptr, dbc, wdbc, Mtok, wdbc, Di, ACT_NONE);
  gemm(s, dbc, wdbc, p.dt_w, Di, p.dt_b, dtb, Di, Mtok, Di, dtr, ACT_SOFTPLUS);
  mamba_scan_kernel<<<cdiv(th, 256), 256, 0, s>>>(dtb, p.A_log, p.D, dbc, wdbc, dtr, dtr + kDS,
                                                  xc, xz, 2 * Di, Di, yb, Bt, L, Di);
  gemm(s, yb, Di, p.out_proj, d, nullptr, out, d, Mtok, d, Di, ACT_NONE);
}

extern "C" void kernel_launch(void* const* d_in, const int* in_sizes, int n_in,
                              void* d_out, int out_size, void* d_ws, size_t ws_size,
                              hipStream_t stream) {
  (void)in_sizes; (void)n_in; (void)out_size; (void)ws_size;
  const float* X    = (const float*)d_in[0];
  const int*   TOD  = (const int*)d_in[1];
  const int*   DOW  = (const int*)d_in[2];
  const int*   SEQ  = (const int*)d_in[3];
  const float* COST = (const float*)d_in[4];
  // params flattened in jax pytree (recursive sorted-key) order, starting at 5:
  //  0 conv1_b  1 conv1_w  2 cost_b  3 cost_w  4 dow_table
  //  5..10  ff_scan {b1,b2,bt,g,w1,w2}
  // 11..16  ff_walk {b1,b2,bt,g,w1,w2}
  // 17 node_table  18 pos_table  19 rg_b1  20 rg_b2  21 rg_w1  22 rg_w2
  // 23 te_b  24 te_w
  // 25..35  temporal_blocks[0] {fwd{A_log,D,conv_b,conv_w,dt_b,dt_w,in_proj,out_proj,x_proj}, ln_b, ln_g}
  // 36..46  temporal_blocks[1]
  // 47..58  tfm_layers[0] {ff1_b,ff1_w,ff2_b,ff2_w,in_b,in_w,ln1_b,ln1_g,ln2_b,ln2_g,out_b,out_w}
  // 59..70  tfm_layers[1]
  // 71 tod_table
  // 72..91  walk_blocks[0] {bwd{9}, fwd{9}, ln_b, ln_g}
  // 92..111 walk_blocks[1]
  auto F = [&](int i) -> const float* { return (const float*)d_in[5 + i]; };
  auto mk = [&](int mb) -> MambaP {
    MambaP p;
    p.A_log = F(mb + 0); p.D = F(mb + 1); p.conv_b = F(mb + 2); p.conv_w = F(mb + 3);
    p.dt_b = F(mb + 4); p.dt_w = F(mb + 5); p.in_proj = F(mb + 6); p.out_proj = F(mb + 7);
    p.x_proj = F(mb + 8);
    return p;
  };

  float* ws = (float*)d_ws;
  size_t off = 0;
  auto alloc = [&](size_t n) -> float* {
    float* p = ws + off;
    off += (n + 63) & ~(size_t)63;
    return p;
  };

  // persistent across phases
  float* h_walk = alloc((size_t)kN * kE);
  float* ht     = alloc((size_t)kB * kN * kT * kDC);
  float* hs     = alloc((size_t)kB * kT * kN * kDC);
  const size_t scratch_base = off;

  // ---------------- phase 1: walk blocks (d=864, Bt=207, L=8) ----------------
  {
    off = scratch_base;
    const int Bt = kN, L = kLW, d = kDW, Di = 2 * kDW;
    const int tok = Bt * L;  // 1656
    float* h   = alloc((size_t)tok * d);
    float* res = alloc((size_t)tok * d);
    float* hn  = alloc((size_t)tok * d);
    float* hnf = alloc((size_t)tok * d);
    float* mo2 = alloc((size_t)tok * d);
    float* xz  = alloc((size_t)tok * 2 * Di);
    float* xc  = alloc((size_t)tok * Di);
    float* dbc = alloc((size_t)tok * 70);
    float* dtb = alloc((size_t)tok * Di);
    float* yb  = alloc((size_t)tok * Di);
    float* hwp = alloc((size_t)kN * kDW);
    float* tw1 = alloc((size_t)kN * kFF);
    float* tw2 = alloc((size_t)kN * kE);

    embed_walk_kernel<<<cdiv(kN * kLW * kTOTW * kE, 256), 256, 0, stream>>>(
        SEQ, COST, F(17), F(3), F(2), F(18), h);

    for (int blk = 0; blk < kNL; ++blk) {
      const int base = 72 + blk * 20;  // bwd@+0, fwd@+9, ln_b@+18, ln_g@+19
      const int nel = tok * d;
      if (blk == 0) copy_kernel<<<cdiv(nel, 256), 256, 0, stream>>>(res, h, nel);
      else          add_kernel<<<cdiv(nel, 256), 256, 0, stream>>>(res, h, nel);
      ln_kernel<<<tok, 256, 0, stream>>>(res, nullptr, F(base + 19), F(base + 18), hn, d);
      // forward direction
      run_mamba(stream, hn, h, xz, xc, dbc, dtb, yb, mk(base + 9), Bt, L, d);
      // backward direction: flip, mamba, flip back, add
      flip_time_kernel<<<cdiv(nel, 256), 256, 0, stream>>>(hn, hnf, Bt, L, d);
      run_mamba(stream, hnf, mo2, xz, xc, dbc, dtb, yb, mk(base + 0), Bt, L, d);
      flip_time_kernel<<<cdiv(nel, 256), 256, 0, stream>>>(mo2, hnf, Bt, L, d);
      add_kernel<<<cdiv(nel, 256), 256, 0, stream>>>(h, hnf, nel);
    }

    // gelu + weighted pool over L, then ff_walk MLP + LN -> h_walk (207 x 32)
    walk_pool_kernel<<<cdiv(kN * kDW, 256), 256, 0, stream>>>(h, F(1), F(0), hwp);
    gemm(stream, hwp, kDW, F(15), kFF, F(11), tw1, kFF, kN, kFF, kDW, ACT_GELU);
    gemm(stream, tw1, kFF, F(16), kE, F(12), tw2, kE, kN, kE, kFF, ACT_NONE);
    ln_kernel<<<kN, 256, 0, stream>>>(tw2, nullptr, F(14), F(13), h_walk, kE);
  }

  // ---------------- phase 2: temporal blocks (d=128, Bt=1656, L=12) ----------------
  {
    off = scratch_base;
    const int Bt = kB * kN, L = kT, d = kDC, Di = 2 * kDC;
    const int tok = Bt * L;  // 19872
    float* res = alloc((size_t)tok * d);
    float* hn  = alloc((size_t)tok * d);
    float* xz  = alloc((size_t)tok * 2 * Di);
    float* xc  = alloc((size_t)tok * Di);
    float* dbc = alloc((size_t)tok * 24);
    float* dtb = alloc((size_t)tok * Di);
    float* yb  = alloc((size_t)tok * Di);

    temporal_embed_kernel<<<cdiv(kB * kT * kN * kDC, 256), 256, 0, stream>>>(
        X, TOD, DOW, F(24), F(23), F(71), F(4), h_walk, ht);

    for (int blk = 0; blk < kNL; ++blk) {
      const int base = 25 + blk * 11;  // fwd@+0, ln_b@+9, ln_g@+10
      const int nel = tok * d;
      if (blk == 0) copy_kernel<<<cdiv(nel, 256), 256, 0, stream>>>(res, ht, nel);
      else          add_kernel<<<cdiv(nel, 256), 256, 0, stream>>>(res, ht, nel);
      ln_kernel<<<tok, 256, 0, stream>>>(res, nullptr, F(base + 10), F(base + 9), hn, d);
      run_mamba(stream, hn, ht, xz, xc, dbc, dtb, yb, mk(base + 0), Bt, L, d);
    }
    ht_to_hs_kernel<<<cdiv(kB * kT * kN * kDC, 256), 256, 0, stream>>>(ht, hs);
  }

  // ---------------- phase 3: transformer layers + heads ----------------
  {
    off = scratch_base;
    const int tok = kB * kT * kN;  // 19872
    float* qkv  = alloc((size_t)tok * 3 * kDC);
    float* att  = alloc((size_t)tok * kDC);
    float* proj = alloc((size_t)tok * kDC);
    float* h1   = alloc((size_t)tok * kDC);
    float* ffb  = alloc((size_t)tok * kFF);
    float* f2   = alloc((size_t)tok * kDC);
    float* g1   = alloc((size_t)tok * kDC);
    float* t1   = alloc((size_t)tok * kFF);
    float* t2   = alloc((size_t)tok * kE);
    float* t3   = alloc((size_t)tok * kE);
    float* r1   = alloc((size_t)tok * kFF);

    for (int l = 0; l < kNL; ++l) {
      const int base = 47 + l * 12;
      gemm(stream, hs, kDC, F(base + 5), 3 * kDC, F(base + 4), qkv, 3 * kDC, tok, 3 * kDC, kDC, ACT_NONE);
      attn_row_kernel<<<kB * kT * kNH * kN, 256, 0, stream>>>(qkv, att);
      gemm(stream, att, kDC, F(base + 11), kDC, F(base + 10), proj, kDC, tok, kDC, kDC, ACT_NONE);
      ln_kernel<<<tok, 256, 0, stream>>>(hs, proj, F(base + 7), F(base + 6), h1, kDC);
      gemm(stream, h1, kDC, F(base + 1), kFF, F(base + 0), ffb, kFF, tok, kFF, kDC, ACT_RELU);
      gemm(stream, ffb, kFF, F(base + 3), kDC, F(base + 2), f2, kDC, tok, kDC, kFF, ACT_NONE);
      ln_kernel<<<tok, 256, 0, stream>>>(h1, f2, F(base + 9), F(base + 8), hs, kDC);
    }

    // h = gelu(hs); ff_scan MLP+LN; gelu; regression head -> d_out (19872 x 1)
    gelu_kernel<<<cdiv(tok * kDC, 256), 256, 0, stream>>>(hs, g1, tok * kDC);
    gemm(stream, g1, kDC, F(9), kFF, F(5), t1, kFF, tok, kFF, kDC, ACT_GELU);
    gemm(stream, t1, kFF, F(10), kE, F(6), t2, kE, tok, kE, kFF, ACT_NONE);
    ln_kernel<<<tok, 256, 0, stream>>>(t2, nullptr, F(8), F(7), t3, kE);
    gelu_kernel<<<cdiv(tok * kE, 256), 256, 0, stream>>>(t3, t2, tok * kE);
    gemm(stream, t2, kE, F(21), kFF, F(19), r1, kFF, tok, kFF, kE, ACT_GELU);
    gemm(stream, r1, kFF, F(22), 1, F(20), (float*)d_out, 1, tok, 1, kFF, ACT_NONE);
  }
}
